// PointerAttention_5265629905043
// MI455X (gfx1250) — matrix operand order repaired and checked by reading, not compiled, not run on hardware
//
#include <hip/hip_runtime.h>
#include <hip/hip_bf16.h>

// Problem sizes (fixed by the reference)
constexpr int kB = 64;
constexpr int kS = 2048;
constexpr int kD = 512;
constexpr int kH = 512;
constexpr int kST = 64;              // S columns per workgroup in the WMMA kernel
constexpr int kLdsStride = kD + 8;   // bf16 units; +8 pads rows to 1040B -> 4-bank
                                     // lane stride -> conflict-free ds_load_b128

typedef __attribute__((ext_vector_type(16))) __bf16 v16bf;
typedef __attribute__((ext_vector_type(8)))  __bf16 v8bf;   // 16B, align 16
typedef __attribute__((ext_vector_type(4)))  __bf16 v4bf;   // 8B,  align 8
typedef __attribute__((ext_vector_type(8)))  float  v8f;

// ---------------------------------------------------------------------------
// helpers
// ---------------------------------------------------------------------------

// A (16x32 bf16) fragment straight from pre-converted bf16 W_ctx:
// this lane holds K = {p[0..7], p[16..23]} (the +half*8 base offset is folded
// into the pointer by the caller). Two 16B global_load_b128, no converts.
__device__ __forceinline__ v16bf load_a_bf16g(const __bf16* __restrict__ p) {
  v8bf lo = *(const v8bf*)p;          // K 0..7   (+half*8)
  v8bf hi = *(const v8bf*)(p + 16);   // K 16..23 (+half*8)
  v16bf r;
#pragma unroll
  for (int i = 0; i < 8; ++i) { r[i] = lo[i]; r[i + 8] = hi[i]; }
  return r;
}

// B (32x16 bf16) fragment from the staged LDS tile: K = {p[0..7], p[16..23]},
// 16B-aligned -> two ds_load_b128.
__device__ __forceinline__ v16bf load_b_lds(const __bf16* p) {
  v8bf lo = *(const v8bf*)p;
  v8bf hi = *(const v8bf*)(p + 16);
  v16bf r;
#pragma unroll
  for (int i = 0; i < 8; ++i) { r[i] = lo[i]; r[i + 8] = hi[i]; }
  return r;
}

__device__ __forceinline__ v8f wmma_bf16(v16bf a, v16bf b, v8f c) {
  // (neg_a, A, neg_b, B, c_mod, C, reuse_a, reuse_b)
  return __builtin_amdgcn_wmma_f32_16x16x32_bf16(false, a, false, b,
                                                 (short)0, c, false, false);
}

// Hardware tanh if the gfx1250 builtin exists; else v_exp_f32-based form that
// saturates to +/-1 for large |x| (no inf/inf NaN).
__device__ __forceinline__ float fast_tanh(float x) {
#if defined(__has_builtin) && __has_builtin(__builtin_amdgcn_tanhf)
  return __builtin_amdgcn_tanhf(x);
#else
  const float e = __expf(2.0f * x);
  return 1.0f - 2.0f / (e + 1.0f);
#endif
}

// ---------------------------------------------------------------------------
// One-time f32 -> bf16 conversion of W_ctx (kH*kD elements, L2-resident).
// ---------------------------------------------------------------------------
__global__ __launch_bounds__(256) void cvt_w_kernel(
    const float* __restrict__ W, __bf16* __restrict__ Wbf) {
  const int i4 = blockIdx.x * 256 + threadIdx.x;   // float4 index
  const float4 g = *(const float4*)(W + (size_t)i4 * 4);
  v4bf pk;
  pk[0] = (__bf16)g.x; pk[1] = (__bf16)g.y;
  pk[2] = (__bf16)g.z; pk[3] = (__bf16)g.w;
  *(v4bf*)(Wbf + (size_t)i4 * 4) = pk;
}

// ---------------------------------------------------------------------------
// out[b,h] = X[b,:] . W[h,:] + bias[h]     (X:[kB,kD], W:[kH,kD])
// ---------------------------------------------------------------------------
__global__ __launch_bounds__(256) void linear_kernel(
    const float* __restrict__ X, const float* __restrict__ W,
    const float* __restrict__ bias, float* __restrict__ out) {
  const int idx = blockIdx.x * 256 + threadIdx.x;   // b*kH + h
  const int b = idx >> 9;
  const int h = idx & (kH - 1);
  const float4* x = (const float4*)(X + (size_t)b * kD);
  const float4* w = (const float4*)(W + (size_t)h * kD);
  float acc = 0.0f;
#pragma unroll 4
  for (int i = 0; i < kD / 4; ++i) {
    float4 xv = x[i], wv = w[i];
    acc = fmaf(xv.x, wv.x, acc);
    acc = fmaf(xv.y, wv.y, acc);
    acc = fmaf(xv.z, wv.z, acc);
    acc = fmaf(xv.w, wv.w, acc);
  }
  out[idx] = acc + bias[h];
}

// ---------------------------------------------------------------------------
// att[b, s0:s0+64] = sum_h V[h] * tanh(inp[b,h] + b_ctx[h] + (W_ctx @ ctx^T))
// 16 waves; wave w owns H rows [32w, 32w+32) x all 64 S cols as 2x4 WMMA
// tiles. B operand (context tile) staged once per workgroup into LDS as bf16;
// A operand streams pre-converted bf16 W_ctx from L2 (no converts in loop).
// ---------------------------------------------------------------------------
__global__ __launch_bounds__(512, 1) void att_kernel(
    const float* __restrict__ context, const __bf16* __restrict__ Wbf,
    const float* __restrict__ b_ctx, const float* __restrict__ V,
    const float* __restrict__ inp, float* __restrict__ att_out) {
  __shared__ __align__(16) __bf16 s_bt[kST * kLdsStride];   // 65 KB bf16 tile
  __shared__ float s_bias[kH];   // inp[b,h] + b_ctx[h]
  __shared__ float s_v[kH];
  __shared__ float s_att[kST];

  const int b   = blockIdx.y;
  const int s0  = blockIdx.x * kST;
  const int tid = threadIdx.x;          // 0..511 == h
  const int lane = tid & 31;
  const int wave = tid >> 5;            // 0..15
  const int half = lane >> 4;           // 0/1
  const int lidx = lane & 15;

  s_bias[tid] = inp[b * kH + tid] + b_ctx[tid];
  s_v[tid]    = V[tid];
  if (tid < kST) s_att[tid] = 0.0f;

  // ---- Stage context[b, s0:s0+64, :] -> bf16 LDS tile (once per block) ----
  // Flat element e = (j*512 + tid)*4: fully coalesced float4 global loads,
  // packed bf16 8B ds_stores, conflict-free (2-bank lane stride).
  {
    const float* tile = context + ((size_t)b * kS + s0) * kD;
#pragma unroll
    for (int j = 0; j < (kST * kD) / (512 * 4); ++j) {   // 16 iterations
      const int e = (j * 512 + tid) * 4;
      const float4 g = *(const float4*)(tile + e);
      const int row = e >> 9;            // e / kD
      const int col = e & (kD - 1);
      v4bf pk;
      pk[0] = (__bf16)g.x; pk[1] = (__bf16)g.y;
      pk[2] = (__bf16)g.z; pk[3] = (__bf16)g.w;
      *(v4bf*)(&s_bt[row * kLdsStride + col]) = pk;
    }
  }
  __syncthreads();

  const int h0 = wave * 32;
  // A-matrix rows: Wbf[h0 + mt*16 + lidx, :], K-swizzled per 16-bit A layout
  const __bf16* a0 = Wbf + (size_t)(h0 + lidx) * kD + half * 8;
  const __bf16* a1 = a0 + (size_t)16 * kD;
  // B-matrix fragments come from the LDS tile: row = nt*16+lidx, K = k0+half*8 (+16)
  const __bf16* bbase = &s_bt[lidx * kLdsStride + half * 8];

  v8f acc[2][4];
  const v8f vzero = {0.f, 0.f, 0.f, 0.f, 0.f, 0.f, 0.f, 0.f};
#pragma unroll
  for (int mt = 0; mt < 2; ++mt)
#pragma unroll
    for (int nt = 0; nt < 4; ++nt) acc[mt][nt] = vzero;

#pragma unroll 2
  for (int k0 = 0; k0 < kD; k0 += 32) {
    v16bf A[2];
    A[0] = load_a_bf16g(a0 + k0);
    A[1] = load_a_bf16g(a1 + k0);
    v16bf Bm[4];
#pragma unroll
    for (int nt = 0; nt < 4; ++nt)
      Bm[nt] = load_b_lds(bbase + nt * 16 * kLdsStride + k0);
#pragma unroll
    for (int mt = 0; mt < 2; ++mt)
#pragma unroll
      for (int nt = 0; nt < 4; ++nt)
        acc[mt][nt] = wmma_bf16(A[mt], Bm[nt], acc[mt][nt]);
  }

  // Epilogue: per C/D layout, VGPR r of lane l holds M = r + 8*(l>=16),
  // N = l&15. Fold V[h]*tanh(ctx + bias) over H into per-column partials.
  float part[4] = {0.f, 0.f, 0.f, 0.f};
#pragma unroll
  for (int mt = 0; mt < 2; ++mt) {
#pragma unroll
    for (int r = 0; r < 8; ++r) {
      const int h = h0 + mt * 16 + half * 8 + r;
      const float bias = s_bias[h];
      const float vv = s_v[h];
#pragma unroll
      for (int nt = 0; nt < 4; ++nt)
        part[nt] += vv * fast_tanh(acc[mt][nt][r] + bias);
    }
  }
#pragma unroll
  for (int nt = 0; nt < 4; ++nt)
    atomicAdd(&s_att[nt * 16 + lidx], part[nt]);   // ds_add_f32
  __syncthreads();

  if (tid < kST) att_out[(size_t)b * kS + s0 + tid] = s_att[tid];
}

// ---------------------------------------------------------------------------
// Masked softmax over S per batch row: alpha[b,:] = softmax(mask ? -inf : att)
// ---------------------------------------------------------------------------
__global__ __launch_bounds__(256) void softmax_kernel(
    const float* __restrict__ att, const unsigned char* __restrict__ mask,
    float* __restrict__ alpha) {
  __shared__ float red[256];
  const int b = blockIdx.x;
  const int tid = threadIdx.x;
  const float NEG_INF = -__builtin_inff();

  float vals[8];
  float m = NEG_INF;
#pragma unroll
  for (int i = 0; i < 8; ++i) {
    const int s = tid + i * 256;
    const float v = mask[b * kS + s] ? NEG_INF : att[(size_t)b * kS + s];
    vals[i] = v;
    m = fmaxf(m, v);
  }
  red[tid] = m;
  __syncthreads();
  for (int off = 128; off > 0; off >>= 1) {
    if (tid < off) red[tid] = fmaxf(red[tid], red[tid + off]);
    __syncthreads();
  }
  m = red[0];
  __syncthreads();

  float sum = 0.0f;
#pragma unroll
  for (int i = 0; i < 8; ++i) {
    const float p = expf(vals[i] - m);
    vals[i] = p;
    sum += p;
  }
  red[tid] = sum;
  __syncthreads();
  for (int off = 128; off > 0; off >>= 1) {
    if (tid < off) red[tid] += red[tid + off];
    __syncthreads();
  }
  const float inv = 1.0f / red[0];
#pragma unroll
  for (int i = 0; i < 8; ++i)
    alpha[(size_t)b * kS + tid + i * 256] = vals[i] * inv;
}

// ---------------------------------------------------------------------------
// cbar[b,d] = sum_s alpha[b,s] * context[b,s,d]   (split S into 8 chunks)
// ---------------------------------------------------------------------------
__global__ __launch_bounds__(512) void cbar_kernel(
    const float* __restrict__ context, const float* __restrict__ alpha,
    float* __restrict__ cbar) {
  const int b = blockIdx.y;
  const int chunk = blockIdx.x;           // 0..7
  const int d = threadIdx.x;              // 0..511
  const int sBeg = chunk * (kS / 8);
  const float* ctx = context + ((size_t)b * kS + sBeg) * kD + d;
  const float* al = alpha + (size_t)b * kS + sBeg;
  float acc = 0.0f;
#pragma unroll 4
  for (int s = 0; s < kS / 8; ++s)
    acc = fmaf(al[s], ctx[(size_t)s * kD], acc);
  atomicAdd(&cbar[b * kD + d], acc);      // global_atomic_add_f32
}

// ---------------------------------------------------------------------------
extern "C" void kernel_launch(void* const* d_in, const int* in_sizes, int n_in,
                              void* d_out, int out_size, void* d_ws,
                              size_t ws_size, hipStream_t stream) {
  const float* input   = (const float*)d_in[0];
  const float* context = (const float*)d_in[1];
  const unsigned char* mask = (const unsigned char*)d_in[2];
  const float* W_in  = (const float*)d_in[3];
  const float* b_in  = (const float*)d_in[4];
  const float* W_ctx = (const float*)d_in[5];
  const float* b_ctx = (const float*)d_in[6];
  const float* V     = (const float*)d_in[7];

  float* out    = (float*)d_out;
  float* hidden = out;                 // [kB*kH]
  float* alpha  = out + kB * kH;       // [kB*kS]

  float*  ws_inp  = (float*)d_ws;              // kB*kH f32
  float*  ws_att  = ws_inp + kB * kH;          // kB*kS f32
  float*  ws_cbar = ws_att + kB * kS;          // kB*kD f32
  __bf16* ws_wbf  = (__bf16*)(ws_cbar + kB * kD);   // kH*kD bf16

  // 0) one-time W_ctx f32 -> bf16 (stays hot in L2 for the big GEMM)
  cvt_w_kernel<<<dim3((kH * kD / 4) / 256), 256, 0, stream>>>(W_ctx, ws_wbf);
  // 1) inp = input @ W_in^T + b_in
  linear_kernel<<<dim3((kB * kH) / 256), 256, 0, stream>>>(input, W_in, b_in,
                                                           ws_inp);
  // 2) att[b,s] = V . tanh(inp + b_ctx + W_ctx @ context^T)   [WMMA bf16]
  att_kernel<<<dim3(kS / kST, kB), 512, 0, stream>>>(context, ws_wbf, b_ctx, V,
                                                     ws_inp, ws_att);
  // 3) alpha = softmax(mask ? -inf : att)
  softmax_kernel<<<kB, 256, 0, stream>>>(ws_att, mask, alpha);
  // 4) cbar[b,:] = sum_s alpha[b,s] * context[b,s,:]
  hipMemsetAsync(ws_cbar, 0, (size_t)kB * kD * sizeof(float), stream);
  cbar_kernel<<<dim3(8, kB), 512, 0, stream>>>(context, alpha, ws_cbar);
  // 5) hidden = cbar @ W_ctx^T + b_ctx   (since sum(alpha)=1)
  linear_kernel<<<dim3((kB * kH) / 256), 256, 0, stream>>>(ws_cbar, W_ctx,
                                                           b_ctx, hidden);
}
